// MultiheadAttention_81458349736092
// MI455X (gfx1250) — compile-verified, matrix operand order checked by reading, not run on hardware
//
#include <hip/hip_runtime.h>
#include <stdint.h>

// Problem constants (match reference)
#define BB 2
#define SS 2048
#define DD 1024
#define HH 16
#define HD 64

typedef __attribute__((ext_vector_type(16))) __bf16 bf16x16;
typedef __attribute__((ext_vector_type(8)))  __bf16 bf16x8;
typedef __attribute__((ext_vector_type(8)))  float  f32x8;

// ---------------------------------------------------------------------------
// Fragment helpers for v_wmma_f32_16x16x32_bf16
// A (16x32, MxK): lane holds row m = lane&15; kbase = (lane>=16)?8:0;
//   elements 0..7 -> K = kbase+e, elements 8..15 -> K = kbase+16+e
// B (32x16, KxN): lane holds col n = lane&15; same per-lane K mapping.
// C/D (16x16 f32): vgpr v holds row m = v + 8*(lane>=16), col n = lane&15.
// ---------------------------------------------------------------------------
__device__ __forceinline__ bf16x16 load_frag16(const __bf16* rowptr, int kbase) {
  bf16x8 lo = *(const bf16x8*)(rowptr + kbase);
  bf16x8 hi = *(const bf16x8*)(rowptr + kbase + 16);
  bf16x16 r;
#pragma unroll
  for (int i = 0; i < 8; ++i) { r[i] = lo[i]; r[i + 8] = hi[i]; }
  return r;
}

__device__ __forceinline__ f32x8 wmma_bf16(bf16x16 a, bf16x16 b, f32x8 c) {
  return __builtin_amdgcn_wmma_f32_16x16x32_bf16(false, a, false, b, (short)0, c,
                                                 false, false);
}

// gfx1250 async global->LDS copy (ASYNCcnt tracked). dst is an LDS byte
// offset (low 32 bits of a generic pointer into __shared__), src is a global
// pointer held in a VGPR pair.
__device__ __forceinline__ void async_copy_b128(uint32_t lds_byte_off, const void* gsrc) {
  asm volatile("global_load_async_to_lds_b128 %0, %1, off"
               :: "v"(lds_byte_off), "v"(gsrc) : "memory");
}
__device__ __forceinline__ void wait_async0() {
  asm volatile("s_wait_asynccnt 0x0" ::: "memory");
}
__device__ __forceinline__ void wait_ds0() {
  asm volatile("s_wait_dscnt 0x0" ::: "memory");
}
__device__ __forceinline__ uint32_t lds_off(const void* p) {
  return (uint32_t)(uintptr_t)p;   // generic LDS pointer: low 32 bits = LDS offset
}

// ---------------------------------------------------------------------------
// Kernel 1: elementwise f32 -> bf16
// ---------------------------------------------------------------------------
__global__ void cvt_f32_bf16(const float* __restrict__ in, __bf16* __restrict__ out, int n) {
  int i = blockIdx.x * blockDim.x + threadIdx.x;
  if (i < n) out[i] = (__bf16)in[i];
}

// ---------------------------------------------------------------------------
// Kernel 2: batched transpose + convert: in[g][r][c] (f32) -> out[g][c][r] (bf16)
// ---------------------------------------------------------------------------
__global__ void transpose_cvt(const float* __restrict__ in, __bf16* __restrict__ out,
                              int rows, int cols) {
  int g = blockIdx.z;
  int idx = blockIdx.x * blockDim.x + threadIdx.x;
  if (idx >= rows * cols) return;
  int r = idx / cols, c = idx % cols;
  size_t base = (size_t)g * rows * cols;
  out[base + (size_t)c * rows + r] = (__bf16)in[base + idx];
}

// ---------------------------------------------------------------------------
// Kernel 3: per-head projection GEMM. One wave computes a 32x64 (rows x HD)
// tile. Ping-pong register double-buffering on the K loop (no copy moves).
// x: [B*S, D] bf16. wT: [H, HD, D] bf16.
// transposed==0 -> out[((b*H+h)*S + s)*HD + e]   (Q, K row-major per head)
// transposed==1 -> out[((b*H+h)*HD + e)*S + s]   (V stored transposed)
// ---------------------------------------------------------------------------
__global__ void __launch_bounds__(32)
proj_kernel(const __bf16* __restrict__ x, const __bf16* __restrict__ wT,
            __bf16* __restrict__ out, int transposed) {
  const int lane = threadIdx.x;
  const int half = lane >> 4;
  const int ln = lane & 15;
  const int kbase = half * 8;
  const int tile = blockIdx.x;   // 32-row tile over B*S
  const int h = blockIdx.y;

  const size_t row0 = (size_t)tile * 32;
  const __bf16* arow0 = x + (row0 + ln) * DD;
  const __bf16* arow1 = x + (row0 + 16 + ln) * DD;
  const __bf16* wh = wT + (size_t)h * HD * DD;

  f32x8 acc[2][4] = {};
  bf16x16 aA[2], bA[4], aB[2], bB[4];

  aA[0] = load_frag16(arow0, kbase);
  aA[1] = load_frag16(arow1, kbase);
#pragma unroll
  for (int t = 0; t < 4; ++t)
    bA[t] = load_frag16(wh + (size_t)(t * 16 + ln) * DD, kbase);

  for (int k0 = 0; k0 < DD; k0 += 64) {   // DD % 64 == 0
    // load set B at k0+32 (always in range: k0 <= DD-64)
    aB[0] = load_frag16(arow0 + k0 + 32, kbase);
    aB[1] = load_frag16(arow1 + k0 + 32, kbase);
#pragma unroll
    for (int t = 0; t < 4; ++t)
      bB[t] = load_frag16(wh + (size_t)(t * 16 + ln) * DD + k0 + 32, kbase);
    // compute on set A
#pragma unroll
    for (int t = 0; t < 4; ++t) {
      acc[0][t] = wmma_bf16(aA[0], bA[t], acc[0][t]);
      acc[1][t] = wmma_bf16(aA[1], bA[t], acc[1][t]);
    }
    // load set A at k0+64
    if (k0 + 64 < DD) {
      aA[0] = load_frag16(arow0 + k0 + 64, kbase);
      aA[1] = load_frag16(arow1 + k0 + 64, kbase);
#pragma unroll
      for (int t = 0; t < 4; ++t)
        bA[t] = load_frag16(wh + (size_t)(t * 16 + ln) * DD + k0 + 64, kbase);
    }
    // compute on set B
#pragma unroll
    for (int t = 0; t < 4; ++t) {
      acc[0][t] = wmma_bf16(aB[0], bB[t], acc[0][t]);
      acc[1][t] = wmma_bf16(aB[1], bB[t], acc[1][t]);
    }
  }

  const int bIdx = (int)(row0 / SS);
  const int s0 = (int)(row0 % SS);
  const size_t head = (size_t)bIdx * HH + h;
#pragma unroll
  for (int r = 0; r < 2; ++r)
#pragma unroll
    for (int t = 0; t < 4; ++t)
#pragma unroll
      for (int v = 0; v < 8; ++v) {
        int m = r * 16 + v + half * 8;
        int n = t * 16 + ln;
        __bf16 val = (__bf16)acc[r][t][v];
        if (!transposed)
          out[(head * SS + s0 + m) * HD + n] = val;
        else
          out[(head * HD + n) * SS + s0 + m] = val;
      }
}

// ---------------------------------------------------------------------------
// Kernel 4: causal flash attention, 32-key tiles.
// 128 threads (4 waves) per (b, h, 64-row q block); wave w owns q-tile qb*4+w.
// Per step, a 32-key K tile (32x64) and V tile (64x32) are staged into
// double-buffered LDS with async global->LDS copies, shared by all 4 waves.
// Score step: 4 QK^T WMMAs; softmax once per 32 keys; P*V uses full K=32.
// Q,K: [B,H,S,HD] bf16.  Vt: [B,H,HD,S] bf16.  O: [B,S,D] bf16 (head concat).
// ---------------------------------------------------------------------------
__global__ void __launch_bounds__(128)
attn_kernel(const __bf16* __restrict__ Q, const __bf16* __restrict__ K,
            const __bf16* __restrict__ Vt, __bf16* __restrict__ O) {
  __shared__ __bf16 ldsK[2][32 * HD];   // [buf][key*HD + hd]    4KB each
  __shared__ __bf16 ldsV[2][HD * 32];   // [buf][hd*32 + key]    4KB each
  __shared__ __bf16 ldsP[4][16 * 32];   // per-wave P transpose  1KB each

  const int tid = threadIdx.x;
  const int wave = tid >> 5;
  const int lane = tid & 31;
  const int half = lane >> 4;
  const int ln = lane & 15;
  const int kbase = half * 8;
  const int qb = blockIdx.x;            // 64-row q block
  const int h = blockIdx.y;
  const int b = blockIdx.z;
  const int qt = qb * 4 + wave;         // this wave's 16-row q tile
  const int ktmax = qb * 2 + 1;         // 32-key tiles needed by this block

  const size_t head = (size_t)b * HH + h;
  const __bf16* Qb = Q + head * SS * HD;
  const __bf16* Kb = K + head * SS * HD;
  const __bf16* Vb = Vt + head * HD * SS;

  // persistent Q fragments for this wave (HD=64 -> two K=32 fragments)
  const bf16x16 aq0 = load_frag16(Qb + (size_t)(qt * 16 + ln) * HD + 0, kbase);
  const bf16x16 aq1 = load_frag16(Qb + (size_t)(qt * 16 + ln) * HD + 32, kbase);

  f32x8 oacc[4] = {};
  float mrow[8], lrow[8];
#pragma unroll
  for (int v = 0; v < 8; ++v) { mrow[v] = -1e30f; lrow[v] = 0.0f; }
  const float scale = 0.125f;  // 1/sqrt(HD)

  // cooperative async stage of one 32-key K tile + V tile (8KB total)
  auto stage = [&](int kt, int buf) {
#pragma unroll
    for (int rep = 0; rep < 2; ++rep) {
      int idx = tid + rep * 128;
      {  // K tile: 32 rows x 64 hd; chunk -> (row idx>>3, 8-elem col (idx&7)*8)
        int r = idx >> 3, c = (idx & 7) * 8;
        async_copy_b128(lds_off(&ldsK[buf][r * HD + c]),
                        Kb + (size_t)(kt * 32 + r) * HD + c);
      }
      {  // V tile: 64 hd rows x 32 keys; chunk -> (row idx>>2, col (idx&3)*8)
        int r = idx >> 2, c = (idx & 3) * 8;
        async_copy_b128(lds_off(&ldsV[buf][r * 32 + c]),
                        Vb + (size_t)r * SS + kt * 32 + c);
      }
    }
  };

  stage(0, 0);
  wait_async0();
  __syncthreads();

  for (int kt = 0; kt <= ktmax; ++kt) {
    const int cur = kt & 1;
    if (kt < ktmax) stage(kt + 1, cur ^ 1);   // prefetch next tile (async)

    if (kt * 32 <= qt * 16 + 15) {  // wave-uniform causal participation
      // S(16x32) = Q * K^T from LDS: two 16x16 sub-tiles
      const __bf16* lk = &ldsK[cur][0];
      bf16x16 bk0lo = load_frag16(lk + (size_t)ln * HD + 0, kbase);
      bf16x16 bk1lo = load_frag16(lk + (size_t)ln * HD + 32, kbase);
      bf16x16 bk0hi = load_frag16(lk + (size_t)(16 + ln) * HD + 0, kbase);
      bf16x16 bk1hi = load_frag16(lk + (size_t)(16 + ln) * HD + 32, kbase);
      f32x8 s0 = {}, s1 = {};
      s0 = wmma_bf16(aq0, bk0lo, s0);
      s0 = wmma_bf16(aq1, bk1lo, s0);
      s1 = wmma_bf16(aq0, bk0hi, s1);
      s1 = wmma_bf16(aq1, bk1hi, s1);

      const int kcol_lo = kt * 32 + ln;
      const int kcol_hi = kcol_lo + 16;
      float alpha[8];
#pragma unroll
      for (int v = 0; v < 8; ++v) {
        const int qrow = qt * 16 + v + half * 8;
        float sv0 = s0[v] * scale;
        float sv1 = s1[v] * scale;
        if (kcol_lo > qrow) sv0 = -1e30f;   // causal mask
        if (kcol_hi > qrow) sv1 = -1e30f;
        float rmax = fmaxf(sv0, sv1);
#pragma unroll
        for (int msk = 1; msk < 16; msk <<= 1)
          rmax = fmaxf(rmax, __shfl_xor(rmax, msk, 32));
        const float mnew = fmaxf(mrow[v], rmax);
        const float p0 = __expf(sv0 - mnew);
        const float p1 = __expf(sv1 - mnew);
        float rsum = p0 + p1;
#pragma unroll
        for (int msk = 1; msk < 16; msk <<= 1)
          rsum += __shfl_xor(rsum, msk, 32);
        alpha[v] = __expf(mrow[v] - mnew);
        lrow[v] = lrow[v] * alpha[v] + rsum;
        mrow[v] = mnew;
        s0[v] = p0;
        s1[v] = p1;
      }
#pragma unroll
      for (int t = 0; t < 4; ++t)
#pragma unroll
        for (int v = 0; v < 8; ++v) oacc[t][v] *= alpha[v];

      // P (16x32): D-layout -> A-fragment layout via wave-private LDS.
      // LDS ops from one wave complete in order; no workgroup barrier needed.
#pragma unroll
      for (int v = 0; v < 8; ++v) {
        const int m = v + half * 8;
        ldsP[wave][m * 32 + ln] = (__bf16)s0[v];
        ldsP[wave][m * 32 + 16 + ln] = (__bf16)s1[v];
      }
      wait_ds0();
      bf16x16 pf = load_frag16(&ldsP[wave][ln * 32], kbase);

      // O += P * V from LDS, full K=32 (B fragment: col n = hd, k = key pos)
#pragma unroll
      for (int t = 0; t < 4; ++t) {
        bf16x16 bv = load_frag16(&ldsV[cur][(t * 16 + ln) * 32], kbase);
        oacc[t] = wmma_bf16(pf, bv, oacc[t]);
      }
    }

    wait_async0();     // this wave's prefetch landed in LDS
    __syncthreads();   // everyone's prefetch landed; safe to read next buffer
  }

  // normalize and store (head-concat layout [B, S, H*HD])
#pragma unroll
  for (int t = 0; t < 4; ++t)
#pragma unroll
    for (int v = 0; v < 8; ++v) {
      const int m = v + half * 8;
      const int n = t * 16 + ln;
      float val = oacc[t][v] / lrow[v];
      O[((size_t)b * SS + qt * 16 + m) * DD + h * HD + n] = (__bf16)val;
    }
}

// ---------------------------------------------------------------------------
// Kernel 5: output projection. out[B*S, D] = X[B*S, D] * Wo via WoT[Dout,Din].
// One wave per 32x64 tile, ping-pong register double-buffering.
// ---------------------------------------------------------------------------
__global__ void __launch_bounds__(32)
outproj_kernel(const __bf16* __restrict__ X, const __bf16* __restrict__ WoT,
               float* __restrict__ out) {
  const int lane = threadIdx.x;
  const int half = lane >> 4;
  const int ln = lane & 15;
  const int kbase = half * 8;
  const int rt = blockIdx.x;   // 32-row tile over B*S
  const int ct = blockIdx.y;   // 64-col tile over D

  const __bf16* arow0 = X + (size_t)(rt * 32 + ln) * DD;
  const __bf16* arow1 = X + (size_t)(rt * 32 + 16 + ln) * DD;
  const __bf16* wc = WoT + (size_t)(ct * 64) * DD;

  f32x8 acc[2][4] = {};
  bf16x16 aA[2], bA[4], aB[2], bB[4];

  aA[0] = load_frag16(arow0, kbase);
  aA[1] = load_frag16(arow1, kbase);
#pragma unroll
  for (int t = 0; t < 4; ++t)
    bA[t] = load_frag16(wc + (size_t)(t * 16 + ln) * DD, kbase);

  for (int k0 = 0; k0 < DD; k0 += 64) {
    aB[0] = load_frag16(arow0 + k0 + 32, kbase);
    aB[1] = load_frag16(arow1 + k0 + 32, kbase);
#pragma unroll
    for (int t = 0; t < 4; ++t)
      bB[t] = load_frag16(wc + (size_t)(t * 16 + ln) * DD + k0 + 32, kbase);
#pragma unroll
    for (int t = 0; t < 4; ++t) {
      acc[0][t] = wmma_bf16(aA[0], bA[t], acc[0][t]);
      acc[1][t] = wmma_bf16(aA[1], bA[t], acc[1][t]);
    }
    if (k0 + 64 < DD) {
      aA[0] = load_frag16(arow0 + k0 + 64, kbase);
      aA[1] = load_frag16(arow1 + k0 + 64, kbase);
#pragma unroll
      for (int t = 0; t < 4; ++t)
        bA[t] = load_frag16(wc + (size_t)(t * 16 + ln) * DD + k0 + 64, kbase);
    }
#pragma unroll
    for (int t = 0; t < 4; ++t) {
      acc[0][t] = wmma_bf16(aB[0], bB[t], acc[0][t]);
      acc[1][t] = wmma_bf16(aB[1], bB[t], acc[1][t]);
    }
  }
#pragma unroll
  for (int r = 0; r < 2; ++r)
#pragma unroll
    for (int t = 0; t < 4; ++t)
#pragma unroll
      for (int v = 0; v < 8; ++v)
        out[(size_t)(rt * 32 + r * 16 + v + half * 8) * DD + ct * 64 + t * 16 + ln] =
            acc[r][t][v];
}

// ---------------------------------------------------------------------------
extern "C" void kernel_launch(void* const* d_in, const int* in_sizes, int n_in,
                              void* d_out, int out_size, void* d_ws, size_t ws_size,
                              hipStream_t stream) {
  const float* q  = (const float*)d_in[0];
  const float* k  = (const float*)d_in[1];
  const float* v  = (const float*)d_in[2];
  const float* Wq = (const float*)d_in[3];
  const float* Wk = (const float*)d_in[4];
  const float* Wv = (const float*)d_in[5];
  const float* Wo = (const float*)d_in[6];
  float* out = (float*)d_out;

  const size_t NX = (size_t)BB * SS * DD;   // activation element count
  const size_t NW = (size_t)HH * DD * HD;   // per-projection weight elements

  __bf16* ws = (__bf16*)d_ws;
  size_t off = 0;
  __bf16* xq  = ws + off; off += NX;
  __bf16* xk  = ws + off; off += NX;
  __bf16* xv  = ws + off; off += NX;
  __bf16* WqT = ws + off; off += NW;
  __bf16* WkT = ws + off; off += NW;
  __bf16* WvT = ws + off; off += NW;
  __bf16* WoT = ws + off; off += (size_t)DD * DD;
  __bf16* Qp  = ws + off; off += NX;   // [B,H,S,HD]
  __bf16* Kp  = ws + off; off += NX;   // [B,H,S,HD]
  __bf16* Vt  = ws + off; off += NX;   // [B,H,HD,S]
  __bf16* AO  = ws + off; off += NX;   // [B,S,D] head-concat attention out
  (void)ws_size; (void)in_sizes; (void)n_in; (void)out_size;

  // 1. convert activations to bf16
  {
    dim3 blk(256), g((unsigned)((NX + 255) / 256));
    cvt_f32_bf16<<<g, blk, 0, stream>>>(q, xq, (int)NX);
    cvt_f32_bf16<<<g, blk, 0, stream>>>(k, xk, (int)NX);
    cvt_f32_bf16<<<g, blk, 0, stream>>>(v, xv, (int)NX);
  }
  // 2. transpose + convert weights
  {
    dim3 blk(256);
    dim3 gw((unsigned)((DD * HD + 255) / 256), 1, HH);
    transpose_cvt<<<gw, blk, 0, stream>>>(Wq, WqT, DD, HD);
    transpose_cvt<<<gw, blk, 0, stream>>>(Wk, WkT, DD, HD);
    transpose_cvt<<<gw, blk, 0, stream>>>(Wv, WvT, DD, HD);
    dim3 go((unsigned)((DD * DD + 255) / 256), 1, 1);
    transpose_cvt<<<go, blk, 0, stream>>>(Wo, WoT, DD, DD);
  }
  // 3. projections (V stored transposed for the P*V stage)
  {
    dim3 pg((unsigned)(BB * SS / 32), HH);
    proj_kernel<<<pg, 32, 0, stream>>>(xq, WqT, Qp, 0);
    proj_kernel<<<pg, 32, 0, stream>>>(xk, WkT, Kp, 0);
    proj_kernel<<<pg, 32, 0, stream>>>(xv, WvT, Vt, 1);
  }
  // 4. causal flash attention (4 waves / block, async double-buffered 32-key tiles)
  attn_kernel<<<dim3(SS / 64, HH, BB), 128, 0, stream>>>(Qp, Kp, Vt, AO);
  // 5. output projection -> f32
  outproj_kernel<<<dim3(BB * SS / 32, DD / 64), 32, 0, stream>>>(AO, WoT, out);
}
